// RelativeTransformer_4930622456118
// MI455X (gfx1250) — compile-verified, matrix operand order checked by reading, not run on hardware
//
#include <hip/hip_runtime.h>

// ---------------------------------------------------------------------------
// Types for gfx1250 WMMA (wave32)
// ---------------------------------------------------------------------------
typedef __bf16 bf16_t;
typedef __attribute__((ext_vector_type(16))) __bf16 v16bf;
typedef __attribute__((ext_vector_type(8)))  float  v8f;

#define H16   16
#define HD64  64
#define L128  128
#define B8    8
#define D1024 1024
#define SCALE 0.125f   // 64^-0.5

// ---------------------------------------------------------------------------
// Load one 16x16x32 bf16 A/B fragment from a row-major matrix (K contiguous).
// Per ISA 7.12.2: lanes 0-15 hold K = {0..7,16..23}, lanes 16-31 K = {8..15,24..31}.
// Works for both global and LDS (generic pointer -> ds_load when shared).
// ---------------------------------------------------------------------------
__device__ __forceinline__ v16bf load_frag(const bf16_t* rowptr, int lane) {
    const int khalf = (lane >> 4) & 1;           // 0 or 1
    const bf16_t* p = rowptr + khalf * 8;        // 16B aligned
    union { v16bf v; uint4 q[2]; } f;
    f.q[0] = *reinterpret_cast<const uint4*>(p);        // K +0..7   (of half)
    f.q[1] = *reinterpret_cast<const uint4*>(p + 16);   // K +16..23 (of half)
    return f.v;
}

__device__ __forceinline__ v8f zero_v8f() {
    v8f z = {0.f, 0.f, 0.f, 0.f, 0.f, 0.f, 0.f, 0.f};
    return z;
}

__device__ __forceinline__ v8f wmma_bf16(v16bf a, v16bf b, v8f c) {
    return __builtin_amdgcn_wmma_f32_16x16x32_bf16(
        /*neg_a=*/false, a, /*neg_b=*/false, b,
        /*c_mod=*/(short)0, c, /*reuse_a=*/false, /*reuse_b=*/false);
}

// ---------------------------------------------------------------------------
// CDNA5 async global->LDS copy (ASYNCcnt-tracked).  ISA §10 / §15.18.3.
// lds_off: per-lane LDS byte offset (flat LDS pointers keep offset in [31:0]).
// gptr:    per-lane 64-bit global address.
// ---------------------------------------------------------------------------
__device__ __forceinline__ void async_load_b128(unsigned lds_off, const void* gptr) {
    asm volatile("global_load_async_to_lds_b128 %0, %1, off"
                 :: "v"(lds_off), "v"(gptr)
                 : "memory");
}

__device__ __forceinline__ void wait_asynccnt0() {
    asm volatile("s_wait_asynccnt 0x0" ::: "memory");
}

// ---------------------------------------------------------------------------
// Kernel 0: fp32 -> bf16 pack, 1M elements per launch, 4 per thread
// ---------------------------------------------------------------------------
__global__ void cvt4_kernel(const float* __restrict__ src, bf16_t* __restrict__ dst) {
    const int i = (blockIdx.x * 256 + threadIdx.x) * 4;
    float4 f = *reinterpret_cast<const float4*>(src + i);
    dst[i + 0] = (bf16_t)f.x;
    dst[i + 1] = (bf16_t)f.y;
    dst[i + 2] = (bf16_t)f.z;
    dst[i + 3] = (bf16_t)f.w;
}

// ---------------------------------------------------------------------------
// Kernel 1: fused q/k/v/r projections.  C = X @ W^T  (1024x1024x1024 each).
//
// Block = 256 threads (8 waves).  Block tile: 128 rows x 64 cols.
//   - Weight tile (64 rows x 32 K, 4 KB) staged into LDS with one
//     global_load_async_to_lds_b128 per thread, double-buffered so the DMA of
//     step k+32 overlaps the WMMAs of step k.  All 8 waves share it.
//   - A fragments (per-wave 16 rows) double-buffered in registers.
// Stores scatter into per-head layouts (see round-0 analysis):
//   gemm 0 (q): new_q = q + s_q -> newq_f32 / newq_bf  [n][l][e]
//   gemm 1 (k): kbf  [n][l][e]
//   gemm 2 (v): vTbf [n][e][l]
//   gemm 3 (r): rstore[n][t][e]   (exact torch reshape-view semantics)
// ---------------------------------------------------------------------------
__global__ void proj_gemm_kernel(const bf16_t* __restrict__ qx,
                                 const bf16_t* __restrict__ px,
                                 const bf16_t* __restrict__ wbf,   // 5 x 1M bf16
                                 const float*  __restrict__ spk_emb,
                                 float*  __restrict__ newq_f32,
                                 bf16_t* __restrict__ newq_bf,
                                 bf16_t* __restrict__ kbf,
                                 bf16_t* __restrict__ vTbf,
                                 float*  __restrict__ rstore) {
    __shared__ bf16_t ldsB[2][64][32];           // 2 x 4 KB weight tiles

    const int wave = threadIdx.x >> 5;           // 0..7
    const int lane = threadIdx.x & 31;
    const int gemm   = blockIdx.x >> 7;          // 0..3
    const int rem    = blockIdx.x & 127;
    const int mg     = rem >> 4;                 // 0..7  (128-row group)
    const int nstrip = rem & 15;                 // 0..15 (64-col strip)
    const int mtile  = mg * 8 + wave;            // 0..63
    const int nbase  = nstrip * 64;

    const bf16_t* Xb = (gemm == 3) ? px : qx;
    const bf16_t* A  = Xb + (size_t)(mtile * 16 + (lane & 15)) * D1024;
    const bf16_t* Wb = wbf + (size_t)gemm * (D1024 * D1024);

    // async-staging assignment: thread -> (weight row, 16B segment)
    const int srow = threadIdx.x >> 2;           // 0..63
    const int sseg = threadIdx.x & 3;            // 0..3

    v8f acc[4];
#pragma unroll
    for (int nt = 0; nt < 4; ++nt) acc[nt] = zero_v8f();

    // prologue: stage K=0 tile, load first A fragment
    async_load_b128((unsigned)(uintptr_t)&ldsB[0][srow][sseg * 8],
                    (const void*)(Wb + (size_t)(nbase + srow) * D1024 + sseg * 8));
    v16bf a_cur = load_frag(A, lane);

    for (int ks = 0; ks < 32; ++ks) {
        const int k   = ks * 32;
        const int buf = ks & 1;

        wait_asynccnt0();        // my slice of tile `buf` landed in LDS
        __syncthreads();         // everyone's slice landed

        if (ks + 1 < 32) {       // kick DMA for next tile (overlaps WMMAs)
            async_load_b128(
                (unsigned)(uintptr_t)&ldsB[buf ^ 1][srow][sseg * 8],
                (const void*)(Wb + (size_t)(nbase + srow) * D1024 + (k + 32) + sseg * 8));
        }
        v16bf a_nxt = a_cur;
        if (ks + 1 < 32) {       // register double-buffer for A
            if (k + 128 < D1024) __builtin_prefetch(A + k + 128, 0, 3);
            a_nxt = load_frag(A + k + 32, lane);
        }

#pragma unroll
        for (int nt = 0; nt < 4; ++nt) {
            v16bf bfrag = load_frag(&ldsB[buf][nt * 16 + (lane & 15)][0], lane);
            acc[nt] = wmma_bf16(a_cur, bfrag, acc[nt]);
        }
        a_cur = a_nxt;
        __syncthreads();         // done reading `buf` before it is re-staged
    }

#pragma unroll
    for (int nt = 0; nt < 4; ++nt) {
#pragma unroll
        for (int r = 0; r < 8; ++r) {
            const int row = r + 8 * (lane >> 4);
            const int gi  = mtile * 16 + row;                 // source row
            const int gj  = nbase + nt * 16 + (lane & 15);    // source col
            const float val = acc[nt][r];
            if (gemm == 0) {            // q : rows = l*8+b
                const int l = gi >> 3, b = gi & 7;
                const int h = gj >> 6, e = gj & 63;
                const int n = b * H16 + h;
                const float nv = val + spk_emb[gj];           // + s_q
                const int idx = (n * L128 + l) * HD64 + e;
                newq_f32[idx] = nv;
                newq_bf[idx]  = (bf16_t)nv;
            } else if (gemm == 1) {     // k
                const int l = gi >> 3, b = gi & 7;
                const int h = gj >> 6, e = gj & 63;
                const int n = b * H16 + h;
                kbf[(n * L128 + l) * HD64 + e] = (bf16_t)val;
            } else if (gemm == 2) {     // v, transposed (n, e, l)
                const int l = gi >> 3, b = gi & 7;
                const int h = gj >> 6, e = gj & 63;
                const int n = b * H16 + h;
                vTbf[(n * HD64 + e) * L128 + l] = (bf16_t)val;
            } else {                    // r : rows = b*128+l (rel_pe layout)
                const int b = gi >> 7, l = gi & 127;
                const int tt = b * 16 + (l >> 3);
                const int n  = (l & 7) * 16 + (gj >> 6);
                const int e  = gj & 63;
                rstore[(n * L128 + tt) * HD64 + e] = val;
            }
        }
    }
}

// ---------------------------------------------------------------------------
// Kernel 2: per-row auxiliary dot products.
//   e0[n,i] = new_q[n,i,:] . spk_emb[0, h*64:..]
//   e1[n,i] = new_q[n,i,:] . spk_emb[1, h*64:..]
//   cvec[n,t] = r[n,t,:]   . spk_emb[0, h*64:..]   (s_q is row-constant)
// ---------------------------------------------------------------------------
__global__ void aux_kernel(const float* __restrict__ newq_f32,
                           const float* __restrict__ rstore,
                           const float* __restrict__ spk_emb,
                           float* __restrict__ e0,
                           float* __restrict__ e1,
                           float* __restrict__ cvec) {
    const int n = blockIdx.x;        // 128
    const int t = threadIdx.x;       // 128
    const int h = n & 15;
    const float* s0 = spk_emb + h * HD64;
    const float* s1 = spk_emb + D1024 + h * HD64;
    const float* qr = newq_f32 + (n * L128 + t) * HD64;
    const float* rr = rstore   + (n * L128 + t) * HD64;
    float a = 0.f, b = 0.f, c = 0.f;
#pragma unroll 8
    for (int e = 0; e < HD64; ++e) {
        const float w0 = s0[e];
        a = fmaf(qr[e], w0,    a);
        b = fmaf(qr[e], s1[e], b);
        c = fmaf(rr[e], w0,    c);
    }
    e0[n * L128 + t]   = a;
    e1[n * L128 + t]   = b;
    cvec[n * L128 + t] = c;
}

// ---------------------------------------------------------------------------
// Kernel 3: attention.  One wave per (n, 16-row tile).
//   scores (WMMA) -> LDS -> bias + mask + softmax (exact 1e-30 semantics)
//   -> probs (bf16) @ v (WMMA) -> ctx_bf in (L,B,D) layout
// ---------------------------------------------------------------------------
__global__ void attn_kernel(const bf16_t* __restrict__ newq_bf,
                            const bf16_t* __restrict__ kbf,
                            const bf16_t* __restrict__ vTbf,
                            const float*  __restrict__ e0,
                            const float*  __restrict__ e1,
                            const float*  __restrict__ cvec,
                            const int*    __restrict__ spk_mask,
                            bf16_t* __restrict__ ctx_bf) {
    __shared__ float sc[16][132];                // 16 rows x 128 cols (+pad)

    const int n     = blockIdx.x >> 3;           // 0..127
    const int itile = blockIdx.x & 7;            // 0..7
    const int lane  = threadIdx.x;               // wave32
    const int b     = n >> 4;
    const int h     = n & 15;

    // ---- stage 1: raw scores a1 = new_q . k^T (K = 64 -> 2 WMMA steps) ----
    const bf16_t* qrow = newq_bf + (n * L128 + itile * 16 + (lane & 15)) * HD64;
    const v16bf a0 = load_frag(qrow,      lane);
    const v16bf a1 = load_frag(qrow + 32, lane);

    for (int jt = 0; jt < 8; ++jt) {
        const bf16_t* krow = kbf + (n * L128 + jt * 16 + (lane & 15)) * HD64;
        const v16bf b0 = load_frag(krow,      lane);
        const v16bf b1 = load_frag(krow + 32, lane);
        v8f acc = zero_v8f();
        acc = wmma_bf16(a0, b0, acc);
        acc = wmma_bf16(a1, b1, acc);
#pragma unroll
        for (int r = 0; r < 8; ++r)
            sc[r + 8 * (lane >> 4)][jt * 16 + (lane & 15)] = acc[r];
    }
    __syncthreads();

    // ---- stage 2: bias terms + mask + softmax, one row at a time ----------
    for (int r = 0; r < 16; ++r) {
        const int i = itile * 16 + r;
        const float e0v = e0[n * L128 + i];
        const float e1v = e1[n * L128 + i];
        float vloc[4];
        float m = -3.0e38f;
#pragma unroll
        for (int u = 0; u < 4; ++u) {
            const int j = lane * 4 + u;
            float s;
            if (j > i) {
                s = 1e-30f;                                   // exact ref value
            } else {
                const float a2v = spk_mask[(b * L128 + i) * L128 + j] ? e1v : e0v;
                const float a3v = cvec[n * L128 + (L128 - 1 - i + j)];
                s = (sc[r][j] + a2v + a3v) * SCALE;
            }
            vloc[u] = s;
            m = fmaxf(m, s);
        }
#pragma unroll
        for (int off = 16; off > 0; off >>= 1)
            m = fmaxf(m, __shfl_xor(m, off, 32));
        float sum = 0.f;
#pragma unroll
        for (int u = 0; u < 4; ++u) {
            vloc[u] = __expf(vloc[u] - m);
            sum += vloc[u];
        }
#pragma unroll
        for (int off = 16; off > 0; off >>= 1)
            sum += __shfl_xor(sum, off, 32);
        const float inv = 1.f / sum;
#pragma unroll
        for (int u = 0; u < 4; ++u)
            sc[r][lane * 4 + u] = vloc[u] * inv;
    }
    __syncthreads();

    // ---- stage 3: out = probs @ v  (K = 128 -> 4 WMMA steps) --------------
    v16bf aw[4];
    {
        const int row   = lane & 15;
        const int khalf = (lane >> 4) & 1;
#pragma unroll
        for (int kt = 0; kt < 4; ++kt) {
            const int kb = kt * 32 + khalf * 8;
            v16bf tf;
#pragma unroll
            for (int u = 0; u < 16; ++u) {
                const int k = kb + ((u >> 3) << 4) + (u & 7);
                tf[u] = (bf16_t)sc[row][k];
            }
            aw[kt] = tf;
        }
    }

#pragma unroll
    for (int et = 0; et < 4; ++et) {
        v8f acc = zero_v8f();
#pragma unroll
        for (int kt = 0; kt < 4; ++kt) {
            const bf16_t* vrow =
                vTbf + (n * HD64 + et * 16 + (lane & 15)) * L128 + kt * 32;
            acc = wmma_bf16(aw[kt], load_frag(vrow, lane), acc);
        }
#pragma unroll
        for (int r = 0; r < 8; ++r) {
            const int i = itile * 16 + r + 8 * (lane >> 4);
            const int e = et * 16 + (lane & 15);
            // ctx matrix row = l*8+b, col = h*64+e   (matches (L,B,D) flat)
            ctx_bf[(i * B8 + b) * D1024 + h * HD64 + e] = (bf16_t)acc[r];
        }
    }
}

// ---------------------------------------------------------------------------
// Kernel 4: final projection  out = ctx @ Wo^T  (fp32 result).
// Same LDS-staged, double-buffered structure as proj_gemm.
// ---------------------------------------------------------------------------
__global__ void final_gemm_kernel(const bf16_t* __restrict__ ctx_bf,
                                  const bf16_t* __restrict__ wo_bf,
                                  float* __restrict__ out) {
    __shared__ bf16_t ldsB[2][64][32];

    const int wave = threadIdx.x >> 5;
    const int lane = threadIdx.x & 31;
    const int mg     = blockIdx.x >> 4;          // 0..7
    const int nstrip = blockIdx.x & 15;          // 0..15
    const int mtile  = mg * 8 + wave;            // 0..63
    const int nbase  = nstrip * 64;

    const bf16_t* A = ctx_bf + (size_t)(mtile * 16 + (lane & 15)) * D1024;

    const int srow = threadIdx.x >> 2;
    const int sseg = threadIdx.x & 3;

    v8f acc[4];
#pragma unroll
    for (int nt = 0; nt < 4; ++nt) acc[nt] = zero_v8f();

    async_load_b128((unsigned)(uintptr_t)&ldsB[0][srow][sseg * 8],
                    (const void*)(wo_bf + (size_t)(nbase + srow) * D1024 + sseg * 8));
    v16bf a_cur = load_frag(A, lane);

    for (int ks = 0; ks < 32; ++ks) {
        const int k   = ks * 32;
        const int buf = ks & 1;

        wait_asynccnt0();
        __syncthreads();

        if (ks + 1 < 32) {
            async_load_b128(
                (unsigned)(uintptr_t)&ldsB[buf ^ 1][srow][sseg * 8],
                (const void*)(wo_bf + (size_t)(nbase + srow) * D1024 + (k + 32) + sseg * 8));
        }
        v16bf a_nxt = a_cur;
        if (ks + 1 < 32) {
            if (k + 128 < D1024) __builtin_prefetch(A + k + 128, 0, 3);
            a_nxt = load_frag(A + k + 32, lane);
        }

#pragma unroll
        for (int nt = 0; nt < 4; ++nt) {
            v16bf bfrag = load_frag(&ldsB[buf][nt * 16 + (lane & 15)][0], lane);
            acc[nt] = wmma_bf16(a_cur, bfrag, acc[nt]);
        }
        a_cur = a_nxt;
        __syncthreads();
    }

#pragma unroll
    for (int nt = 0; nt < 4; ++nt) {
#pragma unroll
        for (int r = 0; r < 8; ++r) {
            const int gi = mtile * 16 + r + 8 * (lane >> 4);
            const int gj = nbase + nt * 16 + (lane & 15);
            out[gi * D1024 + gj] = acc[nt][r];
        }
    }
}

// ---------------------------------------------------------------------------
// Host launcher
// ---------------------------------------------------------------------------
extern "C" void kernel_launch(void* const* d_in, const int* in_sizes, int n_in,
                              void* d_out, int out_size, void* d_ws, size_t ws_size,
                              hipStream_t stream) {
    (void)in_sizes; (void)n_in; (void)out_size; (void)ws_size;

    const float* query   = (const float*)d_in[0];   // (L,B,D)
    const float* rel_pe  = (const float*)d_in[1];   // (B,L,D)
    const float* Wq      = (const float*)d_in[3];
    const float* Wk      = (const float*)d_in[4];
    const float* Wv      = (const float*)d_in[5];
    const float* Wr      = (const float*)d_in[6];
    const float* Wo      = (const float*)d_in[7];
    const float* spk_emb = (const float*)d_in[8];   // (2,D)
    const int*   spk_msk = (const int*)d_in[10];    // (B,L,L)
    float* out = (float*)d_out;

    char* ws = (char*)d_ws;
    const size_t MB = 1u << 20;
    bf16_t* qbf   = (bf16_t*)(ws + 0 * MB);    // 2 MB
    bf16_t* pbf   = (bf16_t*)(ws + 2 * MB);    // 2 MB
    bf16_t* wbf   = (bf16_t*)(ws + 4 * MB);    // 10 MB (Wq,Wk,Wv,Wr,Wo)
    float*  newqf = (float*) (ws + 14 * MB);   // 4 MB
    bf16_t* newqb = (bf16_t*)(ws + 18 * MB);   // 2 MB
    bf16_t* kbf   = (bf16_t*)(ws + 20 * MB);   // 2 MB
    bf16_t* vtb   = (bf16_t*)(ws + 22 * MB);   // 2 MB
    float*  rst   = (float*) (ws + 24 * MB);   // 4 MB
    float*  e0p   = (float*) (ws + 28 * MB);   // 64 KB
    float*  e1p   = (float*) (ws + 28 * MB + 65536);
    float*  cvp   = (float*) (ws + 28 * MB + 131072);
    bf16_t* ctxb  = (bf16_t*)(ws + 29 * MB);   // 2 MB

    const size_t M1 = (size_t)D1024 * D1024;   // 1M elements each

    cvt4_kernel<<<1024, 256, 0, stream>>>(query,  qbf);
    cvt4_kernel<<<1024, 256, 0, stream>>>(rel_pe, pbf);
    cvt4_kernel<<<1024, 256, 0, stream>>>(Wq, wbf + 0 * M1);
    cvt4_kernel<<<1024, 256, 0, stream>>>(Wk, wbf + 1 * M1);
    cvt4_kernel<<<1024, 256, 0, stream>>>(Wv, wbf + 2 * M1);
    cvt4_kernel<<<1024, 256, 0, stream>>>(Wr, wbf + 3 * M1);
    cvt4_kernel<<<1024, 256, 0, stream>>>(Wo, wbf + 4 * M1);

    proj_gemm_kernel<<<512, 256, 0, stream>>>(qbf, pbf, wbf, spk_emb,
                                              newqf, newqb, kbf, vtb, rst);
    aux_kernel<<<128, 128, 0, stream>>>(newqf, rst, spk_emb, e0p, e1p, cvp);
    attn_kernel<<<1024, 32, 0, stream>>>(newqb, kbf, vtb, e0p, e1p, cvp,
                                         spk_msk, ctxb);
    final_gemm_kernel<<<128, 256, 0, stream>>>(ctxb, wbf + 4 * M1, out);
}
